// Attention_57509612094021
// MI455X (gfx1250) — matrix operand order repaired and checked by reading, not compiled, not run on hardware
//
#include <hip/hip_runtime.h>

#define B_    8
#define C_    192
#define HEADS_ 4
#define CPH_  48
#define H_    128
#define W_    128
#define HW_   (H_*W_)
#define CT_   576      // 384 qk channels + 192 v channels
#define CHUNK_ 1024    // split-K chunk for Gram kernel

#define LDS_AS __attribute__((address_space(3)))
#define GLB_AS __attribute__((address_space(1)))

#if defined(__gfx1250__) && __has_builtin(__builtin_amdgcn_global_load_async_to_lds_b32)
#define USE_ASYNC 1
#else
#define USE_ASYNC 0
#endif

typedef __attribute__((ext_vector_type(16))) _Float16 v16h;
typedef __attribute__((ext_vector_type(8)))  float    v8f;

union U16H { uint4 u4[2]; unsigned int u[8]; v16h h; };

static __device__ __forceinline__ v8f wmma_f16(v16h a, v16h b, v8f c) {
  // 8 args: (neg_a, A, neg_b, B, c_mod, C, reuse_a, reuse_b)
  return __builtin_amdgcn_wmma_f32_16x16x32_f16(false, a, false, b, (short)0, c, false, false);
}

// ---------------------------------------------------------------- K0: zero
__global__ void k0_zero(float* __restrict__ p, int n) {
  int i = blockIdx.x * blockDim.x + threadIdx.x;
  if (i < n) p[i] = 0.f;
}

// ------------------------------------------- K1: fused qk+v 1x1 conv (GEMM)
// t[b][co 0..575][n] = Wcat[co][k] * x[b][k][n], f16 output. WMMA over K=192.
__global__ __launch_bounds__(128)
void k1_conv1x1(const float* __restrict__ x, const float* __restrict__ Wqk,
                const float* __restrict__ Wv, _Float16* __restrict__ t)
{
  __shared__ unsigned int lds_u[64 * 97];       // 64 n-rows x 194 halves (pad: conflict-free)
  const int b = blockIdx.z, co0 = blockIdx.y * 64, n0 = blockIdx.x * 64;
  const int tid = threadIdx.x;
  const float* xb = x + (size_t)b * C_ * HW_;
  {  // stage x[k][n-tile] transposed into LDS as [n][k] f16; pack 2 k's per b32 store
    int n = tid & 63;
    int kk0 = (tid >> 6) * 2;                   // 0 or 2
    for (int kk = kk0; kk < C_; kk += 4) {
      union { _Float16 h[2]; unsigned int u; } pk;
      pk.h[0] = (_Float16)xb[(size_t)kk * HW_ + n0 + n];
      pk.h[1] = (_Float16)xb[(size_t)(kk + 1) * HW_ + n0 + n];
      lds_u[n * 97 + (kk >> 1)] = pk.u;
    }
  }
  __syncthreads();
  const int wave = tid >> 5, lane = tid & 31, m = lane & 15;
  const int co_w = co0 + wave * 16;
  const int row  = co_w + m;
  const float* wrow = (row < 384) ? (Wqk + (size_t)row * C_)
                                  : (Wv  + (size_t)(row - 384) * C_);
  const int kbA = (lane < 16) ? 0 : 16;    // A-fragment half-lane K offset
  const int kbBd = (lane < 16) ? 0 : 8;    // B-fragment K offset in dwords (16 halves)
  v16h a[6];
  #pragma unroll
  for (int ks = 0; ks < 6; ++ks) {
    int k0 = ks * 32;
    #pragma unroll
    for (int e = 0; e < 8; ++e) {
      a[ks][e]     = (_Float16)wrow[k0 + kbA + e];
      a[ks][e + 8] = (_Float16)wrow[k0 + kbA + 8 + e];
    }
  }
  v8f acc[4] = {};
  #pragma unroll
  for (int ks = 0; ks < 6; ++ks) {
    int kd = ks * 16 + kbBd;
    #pragma unroll
    for (int j = 0; j < 4; ++j) {
      const unsigned int* lp = &lds_u[(j * 16 + m) * 97 + kd];
      U16H bb;
      #pragma unroll
      for (int q = 0; q < 8; ++q) bb.u[q] = lp[q];
      acc[j] = wmma_f16(a[ks], bb.h, acc[j]);
    }
  }
  const int ro = (lane < 16) ? 0 : 8;
  #pragma unroll
  for (int j = 0; j < 4; ++j) {
    int n = n0 + j * 16 + m;
    #pragma unroll
    for (int r = 0; r < 8; ++r)
      t[((size_t)b * CT_ + co_w + r + ro) * HW_ + n] = (_Float16)acc[j][r];
  }
}

// ---------------------- K2: depthwise 3x3 (+ sum-of-squares for q,k channels)
// q,k channels -> qk[b][ch][n] (channel-major, for Gram); v channels ->
// vt[b][n][cv] (pixel-major, so the output GEMM contracts a contiguous dim).
__global__ __launch_bounds__(256)
void k2_dwconv(const _Float16* __restrict__ t, const float* __restrict__ Wqkdw,
               const float* __restrict__ Wvdw, _Float16* __restrict__ qk,
               _Float16* __restrict__ vt, float* __restrict__ sumsq)
{
  __shared__ _Float16 tile[18 * 132];   // col 1 <-> gc=-1 (zero), col 2+i <-> gc=i
  __shared__ float red[8];
  const int bc = blockIdx.x;                 // b*576 + ch
  const int ch = bc % CT_, b = bc / CT_;
  const int r0 = blockIdx.y * 16;
  const int tid = threadIdx.x;
  const _Float16* src = t + (size_t)bc * HW_;

  for (int r = tid; r < 18; r += 256) {      // halo columns are always zero
    tile[r * 132 + 1]   = (_Float16)0.f;
    tile[r * 132 + 130] = (_Float16)0.f;
  }
#if USE_ASYNC
  for (int p = tid; p < 18 * 64; p += 256) { // interior: async global->LDS b32 copies
    int row = p >> 6, dc = p & 63;
    int gr = r0 + row - 1;
    unsigned int* ldst = (unsigned int*)&tile[row * 132 + 2 + dc * 2];
    if ((unsigned)gr < 128u) {
      void* gp = (void*)(src + gr * W_ + dc * 2);
      void* lp = (void*)ldst;
      __builtin_amdgcn_global_load_async_to_lds_b32((GLB_AS int*)gp, (LDS_AS int*)lp, 0, 0);
    } else {
      *ldst = 0u;
    }
  }
#else
  for (int p = tid; p < 18 * 64; p += 256) {
    int row = p >> 6, dc = p & 63;
    int gr = r0 + row - 1;
    unsigned int vpk = 0u;
    if ((unsigned)gr < 128u) vpk = *(const unsigned int*)(src + gr * W_ + dc * 2);
    *(unsigned int*)&tile[row * 132 + 2 + dc * 2] = vpk;
  }
#endif
  const float* wp = (ch < 384) ? (Wqkdw + ch * 9) : (Wvdw + (ch - 384) * 9);
  float w[9];
  #pragma unroll
  for (int i = 0; i < 9; ++i) w[i] = wp[i];
#if USE_ASYNC
#if __has_builtin(__builtin_amdgcn_s_wait_asynccnt)
  __builtin_amdgcn_s_wait_asynccnt(0);
#else
  asm volatile("s_wait_asynccnt 0x0" ::: "memory");
#endif
#endif
  __syncthreads();
  float ss = 0.f;
  #pragma unroll
  for (int i = 0; i < 8; ++i) {
    int p = tid + i * 256;
    int rr = p >> 7, cc = p & 127;
    float acc = 0.f;
    #pragma unroll
    for (int dy = 0; dy < 3; ++dy)
      #pragma unroll
      for (int dx = 0; dx < 3; ++dx)
        acc += w[dy * 3 + dx] * (float)tile[(rr + dy) * 132 + cc + dx + 1];
    int n = (r0 + rr) * W_ + cc;
    if (ch < 384) qk[((size_t)b * 384 + ch) * HW_ + n] = (_Float16)acc;
    else          vt[((size_t)b * HW_ + n) * C_ + (ch - 384)] = (_Float16)acc;
    ss += acc * acc;
  }
  if (ch < 384) {                       // ch is block-uniform: no divergence hazard
    for (int off = 16; off > 0; off >>= 1) ss += __shfl_down(ss, off, 32);
    if ((tid & 31) == 0) red[tid >> 5] = ss;
    __syncthreads();
    if (tid == 0) {
      float tot = 0.f;
      for (int i = 0; i < 8; ++i) tot += red[i];
      atomicAdd(&sumsq[b * 384 + ch], tot);
    }
  }
}

// -------------------- K3: raw Gram S[b][h] = q_raw . k_raw^T (split-K over n)
// Contraction dim n is contiguous: A and B fragments are direct uint4 loads.
__global__ __launch_bounds__(32)
void k3_gram(const _Float16* __restrict__ qk, float* __restrict__ S)
{
  const int chunk = blockIdx.x, h = blockIdx.y, b = blockIdx.z;
  const int lane = threadIdx.x, m = lane & 15;
  const _Float16* q = qk + ((size_t)b * 384 + h * CPH_) * HW_;
  const _Float16* k = qk + ((size_t)b * 384 + 192 + h * CPH_) * HW_;
  const int n0 = chunk * CHUNK_;
  const int kbA = (lane < 16) ? 0 : 16;
  const int kbB = (lane < 16) ? 0 : 16;
  v8f acc[3][3] = {};
  for (int k0 = 0; k0 < CHUNK_; k0 += 32) {
    v16h a[3], bb[3];
    #pragma unroll
    for (int mt = 0; mt < 3; ++mt) {
      const _Float16* qp = q + (size_t)(mt * 16 + m) * HW_ + n0 + k0;
      const uint4* p4 = (const uint4*)(qp + kbA);
      U16H u; u.u4[0] = p4[0]; u.u4[1] = p4[1];     // K = kb..kb+15 contiguous
      a[mt] = u.h;
      __builtin_prefetch(qp + 256, 0, 1);           // global_prefetch 8 k-steps ahead
    }
    #pragma unroll
    for (int jt = 0; jt < 3; ++jt) {
      const _Float16* kp = k + (size_t)(jt * 16 + m) * HW_ + n0 + k0;
      const uint4* p4 = (const uint4*)(kp + kbB);
      U16H u; u.u4[0] = p4[0]; u.u4[1] = p4[1];     // K = kb..kb+15 contiguous
      bb[jt] = u.h;
      __builtin_prefetch(kp + 256, 0, 1);
    }
    #pragma unroll
    for (int mt = 0; mt < 3; ++mt)
      #pragma unroll
      for (int jt = 0; jt < 3; ++jt)
        acc[mt][jt] = wmma_f16(a[mt], bb[jt], acc[mt][jt]);
  }
  const int ro = (lane < 16) ? 0 : 8;
  float* Sb = S + (size_t)(b * HEADS_ + h) * CPH_ * CPH_;
  #pragma unroll
  for (int mt = 0; mt < 3; ++mt)
    #pragma unroll
    for (int jt = 0; jt < 3; ++jt)
      #pragma unroll
      for (int r = 0; r < 8; ++r)
        atomicAdd(&Sb[(mt * 16 + r + ro) * CPH_ + jt * 16 + m], acc[mt][jt][r]);
}

// ------- K4: normalize + temperature + softmax, fold into projection weights
// M[b][co][h*48+d] = sum_c Wp[co][h*48+c] * attn[b][h][c][d]
__global__ __launch_bounds__(64)
void k4_fold(const float* __restrict__ S, const float* __restrict__ sumsq,
             const float* __restrict__ temp, const float* __restrict__ Wp,
             _Float16* __restrict__ M)
{
  __shared__ float attn[CPH_ * CPH_];
  const int h = blockIdx.x % HEADS_, b = blockIdx.x / HEADS_;
  const int tid = threadIdx.x;
  const float* Sb = S + (size_t)(b * HEADS_ + h) * CPH_ * CPH_;
  const float T = temp[h];
  if (tid < CPH_) {
    int i = tid;
    float qn = fmaxf(sqrtf(fmaxf(sumsq[b * 384 + h * CPH_ + i], 0.f)), 1e-12f);
    float row[CPH_];
    float mx = -3.4e38f;
    for (int j = 0; j < CPH_; ++j) {
      float kn = fmaxf(sqrtf(fmaxf(sumsq[b * 384 + 192 + h * CPH_ + j], 0.f)), 1e-12f);
      float v = Sb[i * CPH_ + j] / (qn * kn) * T;
      row[j] = v; mx = fmaxf(mx, v);
    }
    float s = 0.f;
    for (int j = 0; j < CPH_; ++j) { float e = __expf(row[j] - mx); row[j] = e; s += e; }
    float inv = 1.f / s;
    for (int j = 0; j < CPH_; ++j) attn[i * CPH_ + j] = row[j] * inv;
  }
  __syncthreads();
  for (int idx = tid; idx < C_ * CPH_; idx += 64) {
    int co = idx / CPH_, d = idx % CPH_;
    float a = 0.f;
    for (int c = 0; c < CPH_; ++c)
      a += Wp[(size_t)co * C_ + h * CPH_ + c] * attn[c * CPH_ + d];
    M[((size_t)b * C_ + co) * C_ + h * CPH_ + d] = (_Float16)a;
  }
}

// ------ K5: fused (attn . v + proj): out[b] = M[b](192x192) @ v[b](192xHW)
// vt is pixel-major: B fragments are direct aligned global loads; no LDS.
// Each block: 64 n-cols (one 16-col subtile per wave) x all 192 out channels.
__global__ __launch_bounds__(128)
void k5_out(const _Float16* __restrict__ vt, const _Float16* __restrict__ M,
            float* __restrict__ out)
{
  const int b = blockIdx.z, n0 = blockIdx.x * 64;
  const int tid = threadIdx.x, wave = tid >> 5, lane = tid & 31, m = lane & 15;
  const int nw = n0 + wave * 16;
  const int kbA = (lane < 16) ? 0 : 16;
  const int kbB = (lane < 16) ? 0 : 16;
  const _Float16* vrow = vt + ((size_t)b * HW_ + nw + m) * C_;
  v16h bfr[6];                               // B held in regs, reused for all co
  #pragma unroll
  for (int ks = 0; ks < 6; ++ks) {
    const uint4* p4 = (const uint4*)(vrow + ks * 32 + kbB);
    U16H u; u.u4[0] = p4[0]; u.u4[1] = p4[1];
    bfr[ks] = u.h;
  }
  const int ro = (lane < 16) ? 0 : 8;
  for (int ct = 0; ct < 12; ++ct) {          // 12 x 16 = 192 output channels
    const int co_w = ct * 16;
    const _Float16* mrow = M + ((size_t)b * C_ + co_w + m) * C_;
    v8f acc = {};
    #pragma unroll
    for (int ks = 0; ks < 6; ++ks) {
      const uint4* p4 = (const uint4*)(mrow + ks * 32 + kbA);
      U16H u; u.u4[0] = p4[0]; u.u4[1] = p4[1];
      acc = wmma_f16(u.h, bfr[ks], acc);
    }
    #pragma unroll
    for (int r = 0; r < 8; ++r)
      out[((size_t)b * C_ + co_w + r + ro) * HW_ + nw + m] = acc[r];
  }
}

extern "C" void kernel_launch(void* const* d_in, const int* in_sizes, int n_in,
                              void* d_out, int out_size, void* d_ws, size_t ws_size,
                              hipStream_t stream) {
  (void)in_sizes; (void)n_in; (void)out_size; (void)ws_size;
  const float* x     = (const float*)d_in[0];
  const float* Wqk   = (const float*)d_in[1];
  const float* Wqkdw = (const float*)d_in[2];
  const float* Wv    = (const float*)d_in[3];
  const float* Wvdw  = (const float*)d_in[4];
  const float* Wp    = (const float*)d_in[5];
  const float* temp  = (const float*)d_in[6];

  char* ws = (char*)d_ws;
  size_t off = 0;
  _Float16* t   = (_Float16*)(ws + off); off += (size_t)B_ * CT_ * HW_ * 2;   // ~151 MB
  _Float16* qk  = (_Float16*)(ws + off); off += (size_t)B_ * 384 * HW_ * 2;   // ~101 MB
  _Float16* vt  = (_Float16*)(ws + off); off += (size_t)B_ * HW_ * C_ * 2;    //  ~50 MB
  float* sumsq  = (float*)(ws + off);    off += (size_t)B_ * 384 * 4;
  float* S      = (float*)(ws + off);    off += (size_t)B_ * HEADS_ * CPH_ * CPH_ * 4;
  _Float16* M   = (_Float16*)(ws + off); off += (size_t)B_ * C_ * C_ * 2;
  float* out = (float*)d_out;

  // sumsq and S are contiguous: one zero pass for both atomic targets
  int nz = B_ * 384 + B_ * HEADS_ * CPH_ * CPH_;
  k0_zero<<<(nz + 255) / 256, 256, 0, stream>>>(sumsq, nz);
  k1_conv1x1<<<dim3(HW_ / 64, CT_ / 64, B_), 128, 0, stream>>>(x, Wqk, Wv, t);
  k2_dwconv<<<dim3(B_ * CT_, H_ / 16), 256, 0, stream>>>(t, Wqkdw, Wvdw, qk, vt, sumsq);
  k3_gram<<<dim3(HW_ / CHUNK_, HEADS_, B_), 32, 0, stream>>>(qk, S);
  k4_fold<<<dim3(B_ * HEADS_), 64, 0, stream>>>(S, sumsq, temp, Wp, M);
  k5_out<<<dim3(HW_ / 64, 1, B_), 128, 0, stream>>>(vt, M, out);
}
